// Encoder_83665962926299
// MI455X (gfx1250) — matrix operand order repaired and checked by reading, not compiled, not run on hardware
//
#include <hip/hip_runtime.h>
#include <hip/hip_bf16.h>
#include <math.h>

// ---------------- problem constants (match reference) ----------------
constexpr int N_NODES = 12000;
constexpr int KNBR    = 48;            // neighbors per node == GEMM M rows (3 tiles)
constexpr int CN      = 128;
constexpr int CE      = 128;
constexpr int HID     = 512;
constexpr int EIN     = CN + CE + CN;  // 384
constexpr int NLAY    = 3;
constexpr float MSG_SCALE = 30.0f;
constexpr float LN_EPS    = 1e-5f;

constexpr int NWAVES = 8;              // 256 threads
constexpr int NU_ROWS = 64;            // nodes per workgroup in node-update kernel

typedef __attribute__((ext_vector_type(16))) __bf16 v16bf;
typedef __attribute__((ext_vector_type(8)))  __bf16 v8bf;
typedef __attribute__((ext_vector_type(4)))  __bf16 v4bf;
typedef __attribute__((ext_vector_type(8)))  float  v8f;

union U16bf { v16bf v; v8bf h[2]; };

// Branchless GELU: 0.5x(1+erf(x/sqrt2)); erf via Abramowitz-Stegun 7.1.26
// (max abs err 1.5e-7 << bf16 noise).  v_rcp + v_exp + ~6 fma, no divergence.
__device__ __attribute__((always_inline)) inline float gelu_exact(float x) {
    const float u = x * 0.70710678118654752f;
    const float a = fabsf(u);
    const float t = __frcp_rn(1.0f + 0.3275911f * a);
    float p = 1.061405429f;
    p = p * t - 1.453152027f;
    p = p * t + 1.421413741f;
    p = p * t - 0.284496736f;
    p = p * t + 0.254829592f;
    const float e = 1.0f - (p * t) * __expf(-a * a);
    const float erfv = copysignf(e, u);
    return 0.5f * x * (1.0f + erfv);
}

__device__ __attribute__((always_inline)) inline float wave_sum(float v) {
    #pragma unroll
    for (int off = 16; off >= 1; off >>= 1) v += __shfl_xor(v, off);
    return v;
}

// Fragment layouts per CDNA5 ISA 7.12.2 (wave32):
//   A: lane&15 = M row, lane>>4 = K-half, elems j<8 -> K=kb+(lh*8)+j, j>=8 -> +16
//   B: lane&15 = N col, lane>>4 = K-half, same K order as A (two 16B loads)
//   C: lane&15 = N col, VGPR r -> M = (lane>>4)*8 + r

// -----------------------------------------------------------------------------
// Large-N WMMA GEMM (NP = Nout/32 >= NWAVES): np-outer / kb-mid / mt-inner.
// The two B fragments of an n-pair are loaded ONCE per k-step and feed all
// MTILES accumulator pairs -> MTILES x fewer weight loads, 2*MTILES
// independent WMMA chains in flight.
// -----------------------------------------------------------------------------
template<int MTILES, bool GELU>
__device__ __attribute__((always_inline)) inline
void wgemm_np(const __bf16* As, int lda, int Kin,
              const __bf16* __restrict__ Wg, const float* __restrict__ bg, int Nout,
              __bf16* outb, int ldo, int tid)
{
    const int wave = tid >> 5;
    const int lane = tid & 31;
    const int lh   = lane >> 4;
    const int lm   = lane & 15;
    const int NP   = Nout >> 5;

    for (int np = wave; np < NP; np += NWAVES) {
        const int n0 = np << 5;

        v8f acc0[MTILES], acc1[MTILES];
        #pragma unroll
        for (int mt = 0; mt < MTILES; ++mt)
            #pragma unroll
            for (int r = 0; r < 8; ++r) { acc0[mt][r] = 0.0f; acc1[mt][r] = 0.0f; }

        const __bf16* w0 = Wg + (size_t)(n0 + lm) * Kin + lh * 8;
        const __bf16* w1 = w0 + (size_t)16 * Kin;
        const __bf16* ab = As + (size_t)lm * lda + lh * 8;

        for (int kb = 0; kb < Kin; kb += 32) {
            U16bf b0u, b1u;
            b0u.h[0] = *(const v8bf*)(w0 + kb); b0u.h[1] = *(const v8bf*)(w0 + kb + 16);
            b1u.h[0] = *(const v8bf*)(w1 + kb); b1u.h[1] = *(const v8bf*)(w1 + kb + 16);
            v16bf b0 = b0u.v;
            v16bf b1 = b1u.v;
            #pragma unroll
            for (int mt = 0; mt < MTILES; ++mt) {
                const __bf16* ap = ab + (size_t)(mt * 16) * lda + kb;
                U16bf a;
                a.h[0] = *(const v8bf*)(ap);
                a.h[1] = *(const v8bf*)(ap + 16);
                acc0[mt] = __builtin_amdgcn_wmma_f32_16x16x32_bf16(
                               false, a.v, false, b0, (short)0, acc0[mt], false, false);
                acc1[mt] = __builtin_amdgcn_wmma_f32_16x16x32_bf16(
                               false, a.v, false, b1, (short)0, acc1[mt], false, false);
            }
        }

        const float bias0 = bg[n0 + lm];
        const float bias1 = bg[n0 + 16 + lm];
        #pragma unroll
        for (int mt = 0; mt < MTILES; ++mt) {
            const int mrow = mt * 16 + lh * 8;
            #pragma unroll
            for (int r = 0; r < 8; ++r) {
                float v0 = acc0[mt][r] + bias0;
                float v1 = acc1[mt][r] + bias1;
                if (GELU) { v0 = gelu_exact(v0); v1 = gelu_exact(v1); }
                const size_t rowoff = (size_t)(mrow + r) * ldo;
                outb[rowoff + n0 + lm]      = (__bf16)v0;
                outb[rowoff + n0 + 16 + lm] = (__bf16)v1;
            }
        }
    }
}

// -----------------------------------------------------------------------------
// Small-N WMMA GEMM (pair scheme, tiles distributed over (mt,np)); used when
// NP < NWAVES so np-only distribution would idle waves.
// -----------------------------------------------------------------------------
template<int MTILES, bool GELU, bool F32OUT>
__device__ __attribute__((always_inline)) inline
void wgemm_pair(const __bf16* As, int lda, int Kin,
                const __bf16* __restrict__ Wg, const float* __restrict__ bg, int Nout,
                __bf16* outb, float* outf, int ldo, int tid)
{
    const int wave = tid >> 5;
    const int lane = tid & 31;
    const int lh   = lane >> 4;
    const int lm   = lane & 15;
    const int NP   = Nout >> 5;
    const int T    = MTILES * NP;

    for (int t = wave; t < T; t += NWAVES) {
        const int mt = t % MTILES;
        const int np = t / MTILES;
        const int m0 = mt << 4;
        const int n0 = np << 5;

        v8f acc0, acc1;
        #pragma unroll
        for (int r = 0; r < 8; ++r) { acc0[r] = 0.0f; acc1[r] = 0.0f; }

        const __bf16* abase = As + (size_t)(m0 + lm) * lda + lh * 8;
        const __bf16* w0    = Wg + (size_t)(n0 + lm) * Kin + lh * 8;
        const __bf16* w1    = w0 + (size_t)16 * Kin;

        for (int kb = 0; kb < Kin; kb += 32) {
            U16bf a;
            a.h[0] = *(const v8bf*)(abase + kb);
            a.h[1] = *(const v8bf*)(abase + kb + 16);
            U16bf b0u, b1u;
            b0u.h[0] = *(const v8bf*)(w0 + kb); b0u.h[1] = *(const v8bf*)(w0 + kb + 16);
            b1u.h[0] = *(const v8bf*)(w1 + kb); b1u.h[1] = *(const v8bf*)(w1 + kb + 16);
            v16bf b0 = b0u.v;
            v16bf b1 = b1u.v;
            acc0 = __builtin_amdgcn_wmma_f32_16x16x32_bf16(
                       false, a.v, false, b0, (short)0, acc0, false, false);
            acc1 = __builtin_amdgcn_wmma_f32_16x16x32_bf16(
                       false, a.v, false, b1, (short)0, acc1, false, false);
        }

        const float bias0 = bg[n0 + lm];
        const float bias1 = bg[n0 + 16 + lm];
        const int mrow = m0 + lh * 8;
        #pragma unroll
        for (int r = 0; r < 8; ++r) {
            float v0 = acc0[r] + bias0;
            float v1 = acc1[r] + bias1;
            if (GELU) { v0 = gelu_exact(v0); v1 = gelu_exact(v1); }
            const size_t rowoff = (size_t)(mrow + r) * ldo;
            if (F32OUT) {
                outf[rowoff + n0 + lm]      = v0;
                outf[rowoff + n0 + 16 + lm] = v1;
            } else {
                outb[rowoff + n0 + lm]      = (__bf16)v0;
                outb[rowoff + n0 + 16 + lm] = (__bf16)v1;
            }
        }
    }
}

// -----------------------------------------------------------------------------
// Node-path final stage: msum[n] = sum_over_48_rows( A @ W3^T + b3 )[*, n].
// Chains the 3 M-tiles into ONE accumulator (elementwise C accumulation ==
// row-group sum), then sums 8 regs + shfl_xor(16).  B reused across M-tiles.
// -----------------------------------------------------------------------------
__device__ __attribute__((always_inline)) inline
void wgemm_colsum(const __bf16* As, int lda, int Kin,
                  const __bf16* __restrict__ Wg, const float* __restrict__ bg,
                  float* __restrict__ outg, int tid)
{
    const int wave = tid >> 5;
    const int lane = tid & 31;
    const int lh   = lane >> 4;
    const int lm   = lane & 15;
    const int n0   = wave << 4;            // 8 waves cover 128 cols

    v8f acc;
    #pragma unroll
    for (int r = 0; r < 8; ++r) acc[r] = 0.0f;

    const __bf16* wbase = Wg + (size_t)(n0 + lm) * Kin + lh * 8;
    for (int kb = 0; kb < Kin; kb += 32) {
        U16bf bu;
        bu.h[0] = *(const v8bf*)(wbase + kb); bu.h[1] = *(const v8bf*)(wbase + kb + 16);
        v16bf b = bu.v;
        #pragma unroll
        for (int mt = 0; mt < 3; ++mt) {
            const __bf16* ab = As + (size_t)(mt * 16 + lm) * lda + lh * 8 + kb;
            U16bf a;
            a.h[0] = *(const v8bf*)(ab);
            a.h[1] = *(const v8bf*)(ab + 16);
            acc = __builtin_amdgcn_wmma_f32_16x16x32_bf16(
                      false, a.v, false, b, (short)0, acc, false, false);
        }
    }
    float s = 0.0f;
    #pragma unroll
    for (int r = 0; r < 8; ++r) s += acc[r];
    s += __shfl_xor(s, 16);
    if (lane < 16) outg[n0 + lm] = s + (float)KNBR * bg[n0 + lm];
}

// -----------------------------------------------------------------------------
// Fused per-node message MLP.  MODE 0: node path (writes msum[N,CN]).
//                              MODE 1: edge path (in-place e residual+LN).
// LDS: A(48x384 bf16) | M1(48x512 bf16) | M2(48x512 bf16); MODE1's M3
// (48x128 f32) reuses the A region.
// -----------------------------------------------------------------------------
constexpr size_t MSG_A_BYTES = (size_t)KNBR * EIN * 2;   // 36864
constexpr size_t MSG_M_BYTES = (size_t)KNBR * HID * 2;   // 49152
constexpr size_t MSG_SMEM    = MSG_A_BYTES + 2 * MSG_M_BYTES;  // 135168

template<int MODE>
__global__ __launch_bounds__(256)
void k_msg(const __bf16* __restrict__ hb, const float* __restrict__ ef,
           const int* __restrict__ nbr,
           const __bf16* __restrict__ Wa, const float* __restrict__ ba,
           const __bf16* __restrict__ Wb, const float* __restrict__ bb,
           const __bf16* __restrict__ Wc, const float* __restrict__ bc,
           float* outp,                      // MODE0: msum ; MODE1: e (in/out)
           const float* __restrict__ lng, const float* __restrict__ lnb)
{
    extern __shared__ char smem[];
    __bf16* As = (__bf16*)smem;
    __bf16* M1 = (__bf16*)(smem + MSG_A_BYTES);
    __bf16* M2 = (__bf16*)(smem + MSG_A_BYTES + MSG_M_BYTES);
    float*  M3 = (float*)smem;               // MODE1 only; reuses A region

    const int i    = blockIdx.x;
    const int tid  = threadIdx.x;
    const int wave = tid >> 5;
    const int lane = tid & 31;

    // ---- stage 0: gather [h_i | e_ij | h_j] rows into LDS (16B vectors) ----
    const int ROWV = EIN / 8;                // 48 vec8 slots per row
    for (int s = tid; s < KNBR * ROWV; s += 256) {
        const int row = s / ROWV;
        const int c0  = (s - row * ROWV) * 8;
        v8bf t;
        if (c0 < CN) {
            t = *(const v8bf*)(hb + (size_t)i * CN + c0);
        } else if (c0 < CN + CE) {
            const float* p = ef + ((size_t)i * KNBR + row) * CE + (c0 - CN);
            const float4 f0 = *(const float4*)(p);
            const float4 f1 = *(const float4*)(p + 4);
            t[0]=(__bf16)f0.x; t[1]=(__bf16)f0.y; t[2]=(__bf16)f0.z; t[3]=(__bf16)f0.w;
            t[4]=(__bf16)f1.x; t[5]=(__bf16)f1.y; t[6]=(__bf16)f1.z; t[7]=(__bf16)f1.w;
        } else {
            const int nj = nbr[i * KNBR + row];
            t = *(const v8bf*)(hb + (size_t)nj * CN + (c0 - CN - CE));
        }
        *(v8bf*)(As + (size_t)row * EIN + c0) = t;
    }
    __syncthreads();

    // ---- fused 3-layer MLP ----
    wgemm_np<3, true>(As, EIN, EIN, Wa, ba, HID, M1, HID, tid);
    __syncthreads();
    wgemm_np<3, true>(M1, HID, HID, Wb, bb, HID, M2, HID, tid);
    __syncthreads();

    if (MODE == 0) {
        // register-space row-sum GEMM -> msum, no LDS round-trip
        wgemm_colsum(M2, HID, HID, Wc, bc, outp + (size_t)i * CN, tid);
    } else {
        wgemm_pair<3, false, true>(M2, HID, HID, Wc, bc, CN, nullptr, M3, CN, tid);
        __syncthreads();
        // wave-parallel per-edge residual + LayerNorm (4 cols/lane, shfl reduce)
        for (int row = wave; row < KNBR; row += NWAVES) {
            const size_t ebase = ((size_t)i * KNBR + row) * CE;
            const int c = lane * 4;
            const float4 eo = *(const float4*)(outp + ebase + c);
            const float4 m4 = *(const float4*)(M3 + (size_t)row * CN + c);
            float y0 = eo.x + m4.x, y1 = eo.y + m4.y;
            float y2 = eo.z + m4.z, y3 = eo.w + m4.w;
            const float mean = wave_sum(y0 + y1 + y2 + y3) * (1.0f / CE);
            const float d0 = y0 - mean, d1 = y1 - mean, d2 = y2 - mean, d3 = y3 - mean;
            const float var = wave_sum(d0*d0 + d1*d1 + d2*d2 + d3*d3) * (1.0f / CE);
            const float rs = rsqrtf(var + LN_EPS);
            const float4 g = *(const float4*)(lng + c);
            const float4 b = *(const float4*)(lnb + c);
            float4 o;
            o.x = d0 * rs * g.x + b.x;  o.y = d1 * rs * g.y + b.y;
            o.z = d2 * rs * g.z + b.z;  o.w = d3 * rs * g.w + b.w;
            *(float4*)(outp + ebase + c) = o;
        }
    }
}

// -----------------------------------------------------------------------------
// Node update: u = LN(h + msum/SCALE); d = gelu(u@Wdi^T+bdi)@Wdo^T+bdo;
// h = mask * LN(u + d).  64 nodes / workgroup; both LNs wave-parallel.
// -----------------------------------------------------------------------------
constexpr size_t NU_U_BYTES = (size_t)NU_ROWS * CN * 4;
constexpr size_t NU_A_BYTES = (size_t)NU_ROWS * CN * 2;
constexpr size_t NU_M_BYTES = (size_t)NU_ROWS * HID * 2;
constexpr size_t NU_D_BYTES = (size_t)NU_ROWS * CN * 4;
constexpr size_t NU_SMEM    = NU_U_BYTES + NU_A_BYTES + NU_M_BYTES + NU_D_BYTES;

__global__ __launch_bounds__(256)
void k_node_update(float* __restrict__ hf, __bf16* __restrict__ hb,
                   const float* __restrict__ msum,
                   const float* __restrict__ g1, const float* __restrict__ be1,
                   const __bf16* __restrict__ Wdi, const float* __restrict__ bdi,
                   const __bf16* __restrict__ Wdo, const float* __restrict__ bdo,
                   const float* __restrict__ g2, const float* __restrict__ be2,
                   const float* __restrict__ msk)
{
    extern __shared__ char smem[];
    float*  U = (float*)smem;
    __bf16* A = (__bf16*)(smem + NU_U_BYTES);
    __bf16* M = (__bf16*)(smem + NU_U_BYTES + NU_A_BYTES);
    float*  D = (float*)(smem + NU_U_BYTES + NU_A_BYTES + NU_M_BYTES);

    const int tid  = threadIdx.x;
    const int wave = tid >> 5;
    const int lane = tid & 31;
    const int n0   = blockIdx.x * NU_ROWS;
    const int c    = lane * 4;

    // ---- step 1: u = LN(h + msum/SCALE), wave-parallel per row ----
    for (int r = wave; r < NU_ROWS; r += NWAVES) {
        const int node = n0 + r;
        float y0 = 0, y1 = 0, y2 = 0, y3 = 0;
        if (node < N_NODES) {
            const float4 h4 = *(const float4*)(hf   + (size_t)node * CN + c);
            const float4 m4 = *(const float4*)(msum + (size_t)node * CN + c);
            y0 = h4.x + m4.x * (1.0f / MSG_SCALE);
            y1 = h4.y + m4.y * (1.0f / MSG_SCALE);
            y2 = h4.z + m4.z * (1.0f / MSG_SCALE);
            y3 = h4.w + m4.w * (1.0f / MSG_SCALE);
            const float mean = wave_sum(y0 + y1 + y2 + y3) * (1.0f / CN);
            const float d0 = y0-mean, d1 = y1-mean, d2 = y2-mean, d3 = y3-mean;
            const float var = wave_sum(d0*d0 + d1*d1 + d2*d2 + d3*d3) * (1.0f / CN);
            const float rs = rsqrtf(var + LN_EPS);
            const float4 g = *(const float4*)(g1 + c);
            const float4 b = *(const float4*)(be1 + c);
            y0 = d0*rs*g.x + b.x; y1 = d1*rs*g.y + b.y;
            y2 = d2*rs*g.z + b.z; y3 = d3*rs*g.w + b.w;
        }
        float4 u4; u4.x = y0; u4.y = y1; u4.z = y2; u4.w = y3;
        *(float4*)(U + (size_t)r * CN + c) = u4;
        v4bf a4; a4[0]=(__bf16)y0; a4[1]=(__bf16)y1; a4[2]=(__bf16)y2; a4[3]=(__bf16)y3;
        *(v4bf*)(A + (size_t)r * CN + c) = a4;
    }
    __syncthreads();

    wgemm_np<4, true>(A, CN, CN,  Wdi, bdi, HID, M, HID, tid);
    __syncthreads();
    wgemm_pair<4, false, true>(M, HID, HID, Wdo, bdo, CN, nullptr, D, CN, tid);
    __syncthreads();

    // ---- step 2: h = mask * LN(u + d), wave-parallel per row ----
    for (int r = wave; r < NU_ROWS; r += NWAVES) {
        const int node = n0 + r;
        if (node >= N_NODES) continue;
        const float mk = msk[node];
        const float4 u4 = *(const float4*)(U + (size_t)r * CN + c);
        const float4 d4 = *(const float4*)(D + (size_t)r * CN + c);
        float y0 = u4.x + d4.x, y1 = u4.y + d4.y;
        float y2 = u4.z + d4.z, y3 = u4.w + d4.w;
        const float mean = wave_sum(y0 + y1 + y2 + y3) * (1.0f / CN);
        const float d0 = y0-mean, d1 = y1-mean, d2 = y2-mean, d3 = y3-mean;
        const float var = wave_sum(d0*d0 + d1*d1 + d2*d2 + d3*d3) * (1.0f / CN);
        const float rs = rsqrtf(var + LN_EPS);
        const float4 g = *(const float4*)(g2 + c);
        const float4 b = *(const float4*)(be2 + c);
        y0 = (d0*rs*g.x + b.x) * mk; y1 = (d1*rs*g.y + b.y) * mk;
        y2 = (d2*rs*g.z + b.z) * mk; y3 = (d3*rs*g.w + b.w) * mk;
        float4 o4; o4.x = y0; o4.y = y1; o4.z = y2; o4.w = y3;
        *(float4*)(hf + (size_t)node * CN + c) = o4;
        v4bf h4b; h4b[0]=(__bf16)y0; h4b[1]=(__bf16)y1; h4b[2]=(__bf16)y2; h4b[3]=(__bf16)y3;
        *(v4bf*)(hb + (size_t)node * CN + c) = h4b;
    }
}

// ---------------- small utility kernels ----------------
__global__ void k_cvt_bf16(const float* __restrict__ src, __bf16* __restrict__ dst, size_t n) {
    for (size_t i = blockIdx.x * (size_t)blockDim.x + threadIdx.x; i < n;
         i += (size_t)gridDim.x * blockDim.x)
        dst[i] = (__bf16)src[i];
}

__global__ void k_zero_h(float* __restrict__ hf, __bf16* __restrict__ hb, size_t n) {
    for (size_t i = blockIdx.x * (size_t)blockDim.x + threadIdx.x; i < n;
         i += (size_t)gridDim.x * blockDim.x) {
        hf[i] = 0.0f;
        hb[i] = (__bf16)0.0f;
    }
}

// -----------------------------------------------------------------------------
extern "C" void kernel_launch(void* const* d_in, const int* in_sizes, int n_in,
                              void* d_out, int out_size, void* d_ws, size_t ws_size,
                              hipStream_t stream)
{
    (void)in_sizes; (void)n_in; (void)out_size; (void)ws_size;

    const float* ef_in  = (const float*)d_in[0];
    const int*   nbr    = (const int*)  d_in[1];
    const float* maskp  = (const float*)d_in[2];
    const float* W1f    = (const float*)d_in[3];
    const float* b1f    = (const float*)d_in[4];
    const float* W2f    = (const float*)d_in[5];
    const float* b2f    = (const float*)d_in[6];
    const float* W3f    = (const float*)d_in[7];
    const float* b3f    = (const float*)d_in[8];
    const float* g1f    = (const float*)d_in[9];
    const float* be1f   = (const float*)d_in[10];
    const float* Wdif   = (const float*)d_in[11];
    const float* bdif   = (const float*)d_in[12];
    const float* Wdof   = (const float*)d_in[13];
    const float* bdof   = (const float*)d_in[14];
    const float* g2f    = (const float*)d_in[15];
    const float* be2f   = (const float*)d_in[16];
    const float* W11f   = (const float*)d_in[17];
    const float* b11f   = (const float*)d_in[18];
    const float* W12f   = (const float*)d_in[19];
    const float* b12f   = (const float*)d_in[20];
    const float* W13f   = (const float*)d_in[21];
    const float* b13f   = (const float*)d_in[22];
    const float* g3f    = (const float*)d_in[23];
    const float* be3f   = (const float*)d_in[24];

    // ---- workspace partition ----
    char* ws = (char*)d_ws;
    size_t off = 0;
    auto take = [&](size_t bytes) -> char* {
        char* p = ws + off;
        off = (off + bytes + 255) & ~(size_t)255;
        return p;
    };
    float*  h_f32 = (float*) take((size_t)N_NODES * CN * 4);
    __bf16* h_bf  = (__bf16*)take((size_t)N_NODES * CN * 2);
    float*  msum  = (float*) take((size_t)N_NODES * CN * 4);
    float*  e_f32 = (float*) take((size_t)N_NODES * KNBR * CE * 4);

    const size_t szW1  = (size_t)NLAY * HID * EIN;
    const size_t szW2  = (size_t)NLAY * HID * HID;
    const size_t szW3  = (size_t)NLAY * CN  * HID;
    const size_t szWdi = (size_t)NLAY * HID * CN;
    const size_t szWdo = (size_t)NLAY * CN  * HID;

    __bf16* W1b  = (__bf16*)take(szW1  * 2);
    __bf16* W2b  = (__bf16*)take(szW2  * 2);
    __bf16* W3b  = (__bf16*)take(szW3  * 2);
    __bf16* Wdib = (__bf16*)take(szWdi * 2);
    __bf16* Wdob = (__bf16*)take(szWdo * 2);
    __bf16* W11b = (__bf16*)take(szW1  * 2);
    __bf16* W12b = (__bf16*)take(szW2  * 2);
    __bf16* W13b = (__bf16*)take(szW3  * 2);

    // ---- one-time prep (re-done every call; deterministic) ----
    const int CG = 1024;
    k_cvt_bf16<<<CG, 256, 0, stream>>>(W1f,  W1b,  szW1);
    k_cvt_bf16<<<CG, 256, 0, stream>>>(W2f,  W2b,  szW2);
    k_cvt_bf16<<<CG, 256, 0, stream>>>(W3f,  W3b,  szW3);
    k_cvt_bf16<<<CG, 256, 0, stream>>>(Wdif, Wdib, szWdi);
    k_cvt_bf16<<<CG, 256, 0, stream>>>(Wdof, Wdob, szWdo);
    k_cvt_bf16<<<CG, 256, 0, stream>>>(W11f, W11b, szW1);
    k_cvt_bf16<<<CG, 256, 0, stream>>>(W12f, W12b, szW2);
    k_cvt_bf16<<<CG, 256, 0, stream>>>(W13f, W13b, szW3);
    k_zero_h  <<<CG, 256, 0, stream>>>(h_f32, h_bf, (size_t)N_NODES * CN);
    hipMemcpyAsync(e_f32, ef_in, (size_t)N_NODES * KNBR * CE * sizeof(float),
                   hipMemcpyDeviceToDevice, stream);

    // ---- layer loop ----
    const int nuBlocks = (N_NODES + NU_ROWS - 1) / NU_ROWS;
    for (int l = 0; l < NLAY; ++l) {
        const __bf16* W1l  = W1b  + (size_t)l * HID * EIN;
        const __bf16* W2l  = W2b  + (size_t)l * HID * HID;
        const __bf16* W3l  = W3b  + (size_t)l * CN  * HID;
        const __bf16* Wdil = Wdib + (size_t)l * HID * CN;
        const __bf16* Wdol = Wdob + (size_t)l * CN  * HID;
        const __bf16* W11l = W11b + (size_t)l * HID * EIN;
        const __bf16* W12l = W12b + (size_t)l * HID * HID;
        const __bf16* W13l = W13b + (size_t)l * CN  * HID;

        k_msg<0><<<N_NODES, 256, MSG_SMEM, stream>>>(
            h_bf, e_f32, nbr,
            W1l, b1f + (size_t)l * HID,
            W2l, b2f + (size_t)l * HID,
            W3l, b3f + (size_t)l * CN,
            msum, nullptr, nullptr);

        k_node_update<<<nuBlocks, 256, NU_SMEM, stream>>>(
            h_f32, h_bf, msum,
            g1f + (size_t)l * CN, be1f + (size_t)l * CN,
            Wdil, bdif + (size_t)l * HID,
            Wdol, bdof + (size_t)l * CN,
            g2f + (size_t)l * CN, be2f + (size_t)l * CN,
            maskp);

        k_msg<1><<<N_NODES, 256, MSG_SMEM, stream>>>(
            h_bf, e_f32, nbr,
            W11l, b11f + (size_t)l * HID,
            W12l, b12f + (size_t)l * HID,
            W13l, b13f + (size_t)l * CN,
            e_f32, g3f + (size_t)l * CN, be3f + (size_t)l * CN);
    }

    // ---- outputs: (h, e) concatenated flat, f32 ----
    hipMemcpyAsync(d_out, h_f32, (size_t)N_NODES * CN * sizeof(float),
                   hipMemcpyDeviceToDevice, stream);
    hipMemcpyAsync((char*)d_out + (size_t)N_NODES * CN * sizeof(float),
                   e_f32, (size_t)N_NODES * KNBR * CE * sizeof(float),
                   hipMemcpyDeviceToDevice, stream);
}
